// ConvKanModel_25993142075898
// MI455X (gfx1250) — matrix operand order repaired and checked by reading, not compiled, not run on hardware
//
#include <hip/hip_runtime.h>

#define C_CH 16
#define KB_COUNT 6           // 192 features / 32-K WMMA blocks
#define FEAT_PAD 200         // ushorts per pixel in LDS (192 + pad) -> 400 B stride
#define HALO 18
#define TILE 16
#define IMG 512
#define PLANE (IMG * IMG)
#define WPACK_ELEMS (9 * 6 * 32 * 16)   // taps * kblocks * lanes * 16 bf16
#define NCHUNK 8
#define CHUNK (PLANE / NCHUNK)

typedef __bf16 v16bf __attribute__((ext_vector_type(16)));
typedef __bf16 v8bf  __attribute__((ext_vector_type(8)));
typedef float  v8f   __attribute__((ext_vector_type(8)));

__device__ __forceinline__ unsigned short f2bf(float f) {
    union { float f; unsigned u; } x; x.f = f;
    unsigned r = 0x7FFFu + ((x.u >> 16) & 1u);   // round-to-nearest-even
    return (unsigned short)((x.u + r) >> 16);
}

__device__ __forceinline__ v16bf load_frag16(const unsigned short* p) {
    v8bf lo = *(const v8bf*)(p);
    v8bf hi = *(const v8bf*)(p + 8);
    return __builtin_shufflevector(lo, hi, 0,1,2,3,4,5,6,7,8,9,10,11,12,13,14,15);
}

__device__ __forceinline__ float sigmoidf(float x) { return 1.0f / (1.0f + __expf(-x)); }

// ---------------------------------------------------------------------------
// Pack base_w (16,16,3,3) + spline_w (16,176,3,3) into WMMA A-fragment order.
// Feature index f = c*12 + {0: base(silu), 1+k: spline basis k}.
// A 16x32 bf16 layout: lane half kh, element e -> K = 16*kh + e (matches B contiguous)
// wpack[((tap*6 + kb)*32 + lane)*16 + e]
// ---------------------------------------------------------------------------
__global__ void kan_pack_weights(const float* __restrict__ base_w,
                                 const float* __restrict__ spline_w,
                                 unsigned short* __restrict__ wpack) {
    int i = blockIdx.x * blockDim.x + threadIdx.x;
    if (i >= WPACK_ELEMS) return;
    int e    = i & 15;
    int lane = (i >> 4) & 31;
    int kb   = (i >> 9) % 6;
    int tap  = i / 3072;
    int o  = lane & 15;
    int kh = lane >> 4;
    int k  = kb * 32 + 16 * kh + e;   // feature 0..191
    int c  = k / 12;
    int r  = k - c * 12;
    float w = (r == 0) ? base_w[(o * 16 + c) * 9 + tap]
                       : spline_w[(o * 176 + (c * 11 + (r - 1))) * 9 + tap];
    wpack[i] = f2bf(w);
}

// ---------------------------------------------------------------------------
// Fused KAN conv: features (silu + closed-form cubic B-spline bases, only 4
// nonzero) -> LDS bf16, then implicit-GEMM 3x3 conv via v_wmma_f32_16x16x32_bf16.
// Weights are read from global (identical across WGs -> WGP$); LDS holds only
// the 129.6 KB feature tile so 2 workgroups co-reside per WGP (320 KB LDS).
// MODE 0: raw input. MODE 1: on-the-fly instance-norm + PReLU of prev layer.
// ---------------------------------------------------------------------------
template <int MODE>
__global__ __launch_bounds__(256)
void kan_conv_wmma(const float* __restrict__ in,
                   const float* __restrict__ stats,     // [64][2] mean, rstd
                   const float* __restrict__ prelu_a,   // scalar (MODE 1)
                   const unsigned short* __restrict__ wpack,
                   float* __restrict__ zout) {
    extern __shared__ __align__(16) unsigned short feat[];   // HALO*HALO*FEAT_PAD

    const int tid = threadIdx.x;
    const int bx = blockIdx.x, by = blockIdx.y, b = blockIdx.z;
    const int ox = bx * TILE, oy = by * TILE;
    const float pa = (MODE == 1) ? prelu_a[0] : 0.0f;

    // Warm near caches with the shared packed weights (global_prefetch_b8,
    // locality 3 -> WGP scope: pull into all cache levels for Phase C reuse).
    if (tid < WPACK_ELEMS / 128) {   // 216 cachelines of 256 B
        __builtin_prefetch(wpack + tid * 128, 0, 3);
    }

    // ---- Phase A: halo features into LDS ----
    for (int i = tid; i < C_CH * HALO * HALO; i += 256) {
        int c  = i / (HALO * HALO);
        int rp = i - c * (HALO * HALO);
        int hy = rp / HALO;
        int hx = rp - hy * HALO;
        int gy = oy + hy - 1, gx = ox + hx - 1;
        bool inb = ((unsigned)gy < IMG) && ((unsigned)gx < IMG);

        unsigned int* dst32 = (unsigned int*)(feat + rp * FEAT_PAD + c * 12);
        #pragma unroll
        for (int t = 0; t < 6; t++) dst32[t] = 0u;   // zero feature slot
        unsigned short* dst16 = (unsigned short*)dst32;

        if (inb) {   // conv zero-pads the *feature maps*, so OOB -> all-zero features
            float u = in[(((size_t)b * C_CH + c) * IMG + gy) * IMG + gx];
            float v = u;
            if (MODE == 1) {
                float m  = stats[(b * C_CH + c) * 2 + 0];
                float rs = stats[(b * C_CH + c) * 2 + 1];
                v = (u - m) * rs;
                v = (v >= 0.0f) ? v : pa * v;
            }
            dst16[0] = f2bf(v * sigmoidf(v));   // silu for base conv

            // Uniform cubic B-spline: knots t_j = 0.25*j - 1.75. In knot units
            // s = (v+1.75)/0.25, interval m = floor(s), f = s-m; nonzero bases
            // are indices m-3..m (clipped to 0..10), standard cubic pieces.
            float s = (v + 1.75f) * 4.0f;
            if (s >= 0.0f && s < 14.0f) {
                int   m = (int)s;
                float f = s - (float)m;
                float g = 1.0f - f;
                float b2a = 0.5f * g * g;
                float b2b = 0.5f * ((1.0f + f) * g + (2.0f - f) * f);
                float b2c = 0.5f * f * f;
                const float third = 1.0f / 3.0f;
                float q0 = g * b2a * third;                                // idx m-3
                float q1 = ((f + 2.0f) * b2a + (2.0f - f) * b2b) * third;  // idx m-2
                float q2 = ((f + 1.0f) * b2b + (3.0f - f) * b2c) * third;  // idx m-1
                float q3 = f * b2c * third;                                // idx m
                float q[4] = {q0, q1, q2, q3};
                #pragma unroll
                for (int t = 0; t < 4; t++) {
                    int bi = m - 3 + t;
                    if ((unsigned)bi <= 10u) dst16[1 + bi] = f2bf(q[t]);
                }
            }
        }
    }
    __syncthreads();

    // ---- Phase C: WMMA implicit GEMM, 9 taps x 6 K-blocks, 2 row-tiles ----
    const int wave = tid >> 5;
    const int lane = tid & 31;
    const int n  = lane & 15;   // D column / output pixel column
    const int kh = lane >> 4;   // lane half
    const int r0 = wave;        // output rows handled by this wave
    const int r1 = wave + 8;

    v8f acc0 = {0.f,0.f,0.f,0.f,0.f,0.f,0.f,0.f};
    v8f acc1 = {0.f,0.f,0.f,0.f,0.f,0.f,0.f,0.f};
    #pragma unroll
    for (int ty = 0; ty < 3; ty++) {
        #pragma unroll
        for (int tx = 0; tx < 3; tx++) {
            const unsigned short* ab =
                wpack + ((ty * 3 + tx) * KB_COUNT * 32 + lane) * 16;   // global, cached
            const unsigned short* bb0 =
                feat + ((r0 + ty) * HALO + (n + tx)) * FEAT_PAD + kh * 16;
            const unsigned short* bb1 =
                feat + ((r1 + ty) * HALO + (n + tx)) * FEAT_PAD + kh * 16;
            #pragma unroll
            for (int kb = 0; kb < KB_COUNT; kb++) {
                v16bf af = load_frag16(ab + kb * 512);
                v16bf bf0 = load_frag16(bb0 + kb * 32);
                v16bf bf1 = load_frag16(bb1 + kb * 32);
                acc0 = __builtin_amdgcn_wmma_f32_16x16x32_bf16(
                    false, af, false, bf0, (short)0, acc0, false, false);
                acc1 = __builtin_amdgcn_wmma_f32_16x16x32_bf16(
                    false, af, false, bf1, (short)0, acc1, false, false);
            }
        }
    }

    // D layout: VGPR j, lane L -> channel = j + 8*(L/16), col = L%16
    const int gx = ox + n;
    #pragma unroll
    for (int j = 0; j < 8; j++) {
        int ch = j + 8 * kh;
        size_t base = ((size_t)b * C_CH + ch) * PLANE + (size_t)gx;
        zout[base + (size_t)(oy + r0) * IMG] = acc0[j];
        zout[base + (size_t)(oy + r1) * IMG] = acc1[j];
    }
}

// ---------------------------------------------------------------------------
// Instance-norm stats: 512 partial blocks, then tiny finalize (no atomics).
// ---------------------------------------------------------------------------
__global__ void kan_stats_partial(const float* __restrict__ z,
                                  float* __restrict__ partial) {  // [64*NCHUNK][2]
    __shared__ float ss[256], ss2[256];
    int plane = blockIdx.x;           // 0..63
    int chunk = blockIdx.y;           // 0..NCHUNK-1
    const float* p = z + (size_t)plane * PLANE + (size_t)chunk * CHUNK;
    float s = 0.f, s2 = 0.f;
    for (int i = threadIdx.x; i < CHUNK; i += 256) {
        float v = p[i]; s += v; s2 += v * v;
    }
    ss[threadIdx.x] = s; ss2[threadIdx.x] = s2;
    __syncthreads();
    for (int off = 128; off > 0; off >>= 1) {
        if (threadIdx.x < off) {
            ss[threadIdx.x]  += ss[threadIdx.x + off];
            ss2[threadIdx.x] += ss2[threadIdx.x + off];
        }
        __syncthreads();
    }
    if (threadIdx.x == 0) {
        partial[(plane * NCHUNK + chunk) * 2 + 0] = ss[0];
        partial[(plane * NCHUNK + chunk) * 2 + 1] = ss2[0];
    }
}

__global__ void kan_stats_final(const float* __restrict__ partial,
                                float* __restrict__ stats) {
    int plane = threadIdx.x;
    if (plane >= 64) return;
    float s = 0.f, s2 = 0.f;
    #pragma unroll
    for (int c = 0; c < NCHUNK; c++) {
        s  += partial[(plane * NCHUNK + c) * 2 + 0];
        s2 += partial[(plane * NCHUNK + c) * 2 + 1];
    }
    float m   = s * (1.0f / PLANE);
    float var = s2 * (1.0f / PLANE) - m * m;
    stats[plane * 2 + 0] = m;
    stats[plane * 2 + 1] = rsqrtf(var + 1e-5f);
}

// ---------------------------------------------------------------------------
// Final: instance norm + PReLU + sigmoid.
// ---------------------------------------------------------------------------
__global__ void kan_final(const float* __restrict__ z, const float* __restrict__ stats,
                          const float* __restrict__ prelu_a, float* __restrict__ out) {
    size_t i = (size_t)blockIdx.x * blockDim.x + threadIdx.x;
    if (i >= (size_t)4 * C_CH * PLANE) return;
    int plane = (int)(i / PLANE);
    float m  = stats[plane * 2 + 0];
    float rs = stats[plane * 2 + 1];
    float a  = prelu_a[0];
    float v  = (z[i] - m) * rs;
    v = (v >= 0.f) ? v : a * v;
    out[i] = 1.0f / (1.0f + __expf(-v));
}

// ---------------------------------------------------------------------------
extern "C" void kernel_launch(void* const* d_in, const int* in_sizes, int n_in,
                              void* d_out, int out_size, void* d_ws, size_t ws_size,
                              hipStream_t stream) {
    (void)in_sizes; (void)n_in; (void)out_size; (void)ws_size;
    const float* x   = (const float*)d_in[0];
    const float* bw1 = (const float*)d_in[1];
    const float* sw1 = (const float*)d_in[2];
    const float* a1  = (const float*)d_in[3];
    const float* bw2 = (const float*)d_in[4];
    const float* sw2 = (const float*)d_in[5];
    const float* a2  = (const float*)d_in[6];
    float* out = (float*)d_out;

    char* ws = (char*)d_ws;
    const size_t zbytes = (size_t)4 * C_CH * PLANE * sizeof(float);   // 67,108,864
    float* z1  = (float*)(ws);
    float* z2  = (float*)(ws + zbytes);
    float* st1 = (float*)(ws + 2 * zbytes);
    float* st2 = (float*)(ws + 2 * zbytes + 1024);
    float* prt = (float*)(ws + 2 * zbytes + 2048);                    // 4 KB partials
    unsigned short* wp1 = (unsigned short*)(ws + 2 * zbytes + 6144);
    unsigned short* wp2 = (unsigned short*)(ws + 2 * zbytes + 6144 + WPACK_ELEMS * 2);

    const int shbytes = HALO * HALO * FEAT_PAD * 2;   // 129,600 B -> 2 WGs/WGP
    hipFuncSetAttribute(reinterpret_cast<const void*>(kan_conv_wmma<0>),
                        hipFuncAttributeMaxDynamicSharedMemorySize, shbytes);
    hipFuncSetAttribute(reinterpret_cast<const void*>(kan_conv_wmma<1>),
                        hipFuncAttributeMaxDynamicSharedMemorySize, shbytes);

    dim3 cgrid(IMG / TILE, IMG / TILE, 4);   // 32 x 32 x 4
    dim3 sgrid(64, NCHUNK);

    kan_pack_weights<<<WPACK_ELEMS / 256, 256, 0, stream>>>(bw1, sw1, wp1);
    kan_pack_weights<<<WPACK_ELEMS / 256, 256, 0, stream>>>(bw2, sw2, wp2);

    kan_conv_wmma<0><<<cgrid, 256, shbytes, stream>>>(x, nullptr, nullptr, wp1, z1);
    kan_stats_partial<<<sgrid, 256, 0, stream>>>(z1, prt);
    kan_stats_final<<<1, 64, 0, stream>>>(prt, st1);

    kan_conv_wmma<1><<<cgrid, 256, shbytes, stream>>>(z1, st1, a1, wp2, z2);
    kan_stats_partial<<<sgrid, 256, 0, stream>>>(z2, prt);
    kan_stats_final<<<1, 64, 0, stream>>>(prt, st2);

    kan_final<<<(4 * C_CH * PLANE) / 256, 256, 0, stream>>>(z2, st2, a2, out);
}
